// DGLlifeGCNClassifier_11785390260439
// MI455X (gfx1250) — compile-verified, matrix operand order checked by reading, not run on hardware
//
#include <hip/hip_runtime.h>

// ---------------------------------------------------------------------------
// DGL-life GCN classifier for MI455X (gfx1250, wave32, WMMA).
// Dense GEMMs: v_wmma_f32_16x16x32_f16, fragments staged in LDS in the native
// fragment layout so lanes read them with ds_load_b128.
// Edge aggregation: native f32 global atomics (node matrix is L2-resident:
// 128MB < 192MB L2).
// ---------------------------------------------------------------------------

typedef __attribute__((ext_vector_type(16))) _Float16 v16h;
typedef __attribute__((ext_vector_type(8)))  float    v8f;

#define N_NODES  500000
#define N_EDGES  8000000
#define N_GRAPHS 16384
#define IN_FEATS 74
#define HID      64
#define CLS_HID  128

// ---------------------------------------------------------------------------
// Tiled WMMA GEMM:  out = epilogue( A[M,K] @ W[K, ldw] ), K <= 128.
// Block = 256 threads = 8 waves laid out as 2 (M) x 4 (N) 16x16 tiles.
// Block covers 64 output columns (gridDim.y selects the 64-col chunk) and
// 256 rows via an 8-iteration row-tile loop (weights staged to LDS once).
// LDS holds A and W pre-swizzled into the ISA 7.12.2 16-bit fragment layout:
//   k -> (g = k>>3): lane = m|n + ((g&1)<<4), half = (k&7) + ((g&2)?8:0)
// so each lane loads its v16h fragment as two ds_load_b128.
// mode 0: out = acc
// mode 1: out = relu(acc + bias)
// mode 2: out = relu(acc + bias) + relu(aux + bias2)   (aux may alias out)
// ---------------------------------------------------------------------------
__global__ __launch_bounds__(256)
void gemm_wmma(const float* __restrict__ A, int M, int K, int lda,
               const float* __restrict__ W, int ldw,
               const float* __restrict__ bias, const float* __restrict__ bias2,
               const float* __restrict__ aux,
               float* __restrict__ out, int ldo, int mode)
{
    __shared__ v16h sWf[4 * 4 * 32];   // [ntile][kstep][lane] fragments, 16 KB
    __shared__ v16h sAf[2 * 4 * 32];   // [mtile][kstep][lane] fragments,  8 KB

    const int tid    = threadIdx.x;
    const int wave   = tid >> 5;
    const int lane   = tid & 31;
    const int mtile  = wave >> 2;          // 0..1
    const int ntile  = wave & 3;           // 0..3
    const int n      = lane & 15;
    const int mhi    = (lane >> 4) * 8;    // C/D row offset for upper lanes
    const int ksteps = (K + 31) >> 5;      // <= 4
    const int Kpad   = ksteps << 5;
    const int cchunk = blockIdx.y * 64;

    // ---- Stage weight chunk once, pre-swizzled to fragment layout ----------
    for (int idx = tid; idx < Kpad * 64; idx += 256) {
        const int k = idx >> 6, c = idx & 63;
        const int kc = (k < K) ? k : (K - 1);
        float v = W[(size_t)kc * ldw + cchunk + c];
        v = (k < K) ? v : 0.0f;
        const int nt = c >> 4, nn = c & 15;
        const int ks = k >> 5, kk = k & 31, g = kk >> 3;
        const int ln = nn + ((g & 1) << 4);
        const int hi = (kk & 7) + ((g & 2) ? 8 : 0);
        ((_Float16*)sWf)[((((nt << 2) + ks) << 5) + ln) * 16 + hi] = (_Float16)v;
    }
    __syncthreads();

    const int kchunk = Kpad >> 3;          // consecutive k per staging thread (mult of 4)
    const int col    = cchunk + ntile * 16 + n;
    const float b1   = (mode >= 1) ? bias[col]  : 0.0f;
    const float b2   = (mode == 2) ? bias2[col] : 0.0f;

    for (int rt = 0; rt < 8; ++rt) {
        const int rowbase = blockIdx.x * 256 + rt * 32;

        // ---- Stage 32 rows x Kpad of A, pre-swizzled (branch-free) ---------
        {
            const int r32  = tid >> 3;             // 0..31
            const int grow = rowbase + r32;
            const int grc  = (grow < M) ? grow : (M - 1);
            const int mt   = r32 >> 4, m = r32 & 15;
            const int koff = (tid & 7) * kchunk;
#pragma unroll 4
            for (int kb0 = 0; kb0 < kchunk; kb0 += 4) {
                const int kb = koff + kb0;
                const int ks = kb >> 5, kk = kb & 31, g = kk >> 3;
                const int ln = m + ((g & 1) << 4);
                const int ib = (kk & 7) + ((g & 2) ? 8 : 0);
                _Float16* dst =
                    ((_Float16*)sAf) + ((((mt << 2) + ks) << 5) + ln) * 16 + ib;
#pragma unroll
                for (int j = 0; j < 4; ++j) {
                    const int gk = kb + j;
                    const int gkc = (gk < K) ? gk : (K - 1);
                    float v = A[(size_t)grc * lda + gkc];
                    v = (grow < M && gk < K) ? v : 0.0f;
                    dst[j] = (_Float16)v;
                }
            }
        }
        __syncthreads();

        // ---- WMMA over K: fragments are contiguous v16h -> ds_load_b128 ----
        v8f acc = {};
        for (int ks = 0; ks < ksteps; ++ks) {
            const v16h a = sAf[(((mtile << 2) + ks) << 5) + lane];
            const v16h b = sWf[(((ntile << 2) + ks) << 5) + lane];
            acc = __builtin_amdgcn_wmma_f32_16x16x32_f16(
                      false, a, false, b, (short)0, acc, false, false);
        }

        // ---- Epilogue + store ---------------------------------------------
#pragma unroll
        for (int r = 0; r < 8; ++r) {
            const int row = rowbase + mtile * 16 + mhi + r;
            if (row < M) {
                float v = acc[r];
                if (mode >= 1) v = fmaxf(v + b1, 0.0f);
                if (mode == 2) v += fmaxf(aux[(size_t)row * ldo + col] + b2, 0.0f);
                out[(size_t)row * ldo + col] = v;
            }
        }
        __syncthreads();   // protect sAf before next row-tile restage
    }
}

// ---------------------------------------------------------------------------
// Edge scatter: one wave32 per edge, lane handles 2 of 64 features.
// agg[dst] += t[src]  (f32 atomics; working set is L2-resident).
// ---------------------------------------------------------------------------
__global__ __launch_bounds__(256)
void edge_scatter(const int* __restrict__ src, const int* __restrict__ dst,
                  const float* __restrict__ t, float* __restrict__ agg,
                  int nedges)
{
    const int edge = blockIdx.x * 8 + (threadIdx.x >> 5);
    if (edge >= nedges) return;
    const int lane = threadIdx.x & 31;
    const int s = src[edge], d = dst[edge];
    const float2 v = *reinterpret_cast<const float2*>(t + (size_t)s * 64 + lane * 2);
    float* ap = agg + (size_t)d * 64 + lane * 2;
    unsafeAtomicAdd(ap + 0, v.x);
    unsafeAtomicAdd(ap + 1, v.y);
}

// ---------------------------------------------------------------------------
__global__ void zero_f4(float4* __restrict__ p, int n4)
{
    const int i = blockIdx.x * blockDim.x + threadIdx.x;
    if (i < n4) p[i] = make_float4(0.f, 0.f, 0.f, 0.f);
}

// Sum-pool nodes -> graphs (graph_ids sorted; atomics into 4MB table).
__global__ void pool_sum(const float* __restrict__ h, const int* __restrict__ gid,
                         float* __restrict__ gfeat, int nnodes)
{
    const int idx = blockIdx.x * blockDim.x + threadIdx.x;
    if (idx >= nnodes * 64) return;
    const int node = idx >> 6, f = idx & 63;
    unsafeAtomicAdd(&gfeat[(size_t)gid[node] * 64 + f], h[idx]);
}

// logits[g,c] = hidden[g,:] . cW2[:,c] + cb2[c]   (K=128, 2 classes)
__global__ void cls_head(const float* __restrict__ hidden,
                         const float* __restrict__ cW2,
                         const float* __restrict__ cb2,
                         float* __restrict__ logits, int ngraphs)
{
    const int idx = blockIdx.x * blockDim.x + threadIdx.x;
    if (idx >= ngraphs * 2) return;
    const int g = idx >> 1, c = idx & 1;
    const float* hp = hidden + (size_t)g * CLS_HID;
    float acc = cb2[c];
#pragma unroll 8
    for (int k = 0; k < CLS_HID; ++k) acc += hp[k] * cW2[k * 2 + c];
    logits[idx] = acc;
}

// ---------------------------------------------------------------------------
extern "C" void kernel_launch(void* const* d_in, const int* in_sizes, int n_in,
                              void* d_out, int out_size, void* d_ws, size_t ws_size,
                              hipStream_t stream)
{
    const float* x    = (const float*)d_in[0];
    const int*   src  = (const int*)d_in[1];
    const int*   dst  = (const int*)d_in[2];
    const int*   gid  = (const int*)d_in[3];
    const float* gW[3] = { (const float*)d_in[4],  (const float*)d_in[8],  (const float*)d_in[12] };
    const float* gb[3] = { (const float*)d_in[5],  (const float*)d_in[9],  (const float*)d_in[13] };
    const float* rW[3] = { (const float*)d_in[6],  (const float*)d_in[10], (const float*)d_in[14] };
    const float* rb[3] = { (const float*)d_in[7],  (const float*)d_in[11], (const float*)d_in[15] };
    const float* cW1 = (const float*)d_in[16];
    const float* cb1 = (const float*)d_in[17];
    const float* cW2 = (const float*)d_in[18];
    const float* cb2 = (const float*)d_in[19];

    const size_t n64 = (size_t)N_NODES * 64;
    float* buf0   = (float*)d_ws;                    // t (projected features)
    float* buf1   = buf0 + n64;                      // agg / h (ping)
    float* buf2   = buf1 + n64;                      // agg / h (pong)
    float* gfeat  = buf2 + n64;                      // 16384 x 64
    float* hidden = gfeat + (size_t)N_GRAPHS * 64;   // 16384 x 128

    const int mblk = (N_NODES + 255) / 256;          // 1954 row blocks
    const int eblk = (N_EDGES + 7) / 8;              // 1M blocks, 8 edges each
    const int zblk = (int)((n64 / 4 + 255) / 256);

    // Layer schedule: fused epilogue overwrites agg -> h_out.
    const float* h = x;
    float* aggbufs[3] = { buf1, buf2, buf1 };
    int kin = IN_FEATS;

    for (int L = 0; L < 3; ++L) {
        float* agg = aggbufs[L];
        // t = h @ gW[L]   (project first: segsum(h[src])@gW == segsum((h@gW)[src]))
        gemm_wmma<<<dim3(mblk, 1), 256, 0, stream>>>(
            h, N_NODES, kin, kin, gW[L], HID,
            nullptr, nullptr, nullptr, buf0, HID, 0);
        // agg = 0
        zero_f4<<<zblk, 256, 0, stream>>>((float4*)agg, (int)(n64 / 4));
        // agg[dst] += t[src]
        edge_scatter<<<eblk, 256, 0, stream>>>(src, dst, buf0, agg, N_EDGES);
        // h_out = relu(h @ rW + rb) + relu(agg + gb)   (in place over agg)
        gemm_wmma<<<dim3(mblk, 1), 256, 0, stream>>>(
            h, N_NODES, kin, kin, rW[L], HID,
            rb[L], gb[L], agg, agg, HID, 2);
        h = agg;
        kin = HID;
    }

    // Sum-pool readout.
    const int gz = (N_GRAPHS * 64 / 4 + 255) / 256;
    zero_f4<<<gz, 256, 0, stream>>>((float4*)gfeat, N_GRAPHS * 64 / 4);
    pool_sum<<<(N_NODES * 64 + 255) / 256, 256, 0, stream>>>(h, gid, gfeat, N_NODES);

    // hidden = relu(gfeat @ cW1 + cb1)   (M=16384, K=64, N=128)
    gemm_wmma<<<dim3((N_GRAPHS + 255) / 256, 2), 256, 0, stream>>>(
        gfeat, N_GRAPHS, HID, HID, cW1, CLS_HID,
        cb1, nullptr, nullptr, hidden, CLS_HID, 1);

    // logits = hidden @ cW2 + cb2
    cls_head<<<(N_GRAPHS * 2 + 255) / 256, 256, 0, stream>>>(
        hidden, cW2, cb2, (float*)d_out, N_GRAPHS);

    (void)in_sizes; (void)n_in; (void)out_size; (void)ws_size;
}
